// BiMamba2Layer_26173530702492
// MI455X (gfx1250) — compile-verified, matrix operand order checked by reading, not run on hardware
//
#include <hip/hip_runtime.h>
#include <hip/hip_bf16.h>
#include <math.h>
#include <stdint.h>

typedef __attribute__((ext_vector_type(16))) __bf16 v16bf;
typedef __attribute__((ext_vector_type(8)))  float  v8f;

constexpr int kB  = 4;
constexpr int kL  = 2048;
constexpr int kDM = 1024;   // D_MODEL
constexpr int kDS = 64;     // D_STATE
constexpr int kDC = 4;      // D_CONV
constexpr int kDI = 2048;   // D_INNER
constexpr int kNH = 32;     // NHEADS
constexpr int kHD = 64;     // HEADDIM
constexpr int kCD = 2176;   // CONV_DIM = D_INNER + 2*D_STATE
constexpr int kDIP = 4256;  // D_IN_PROJ = 2*D_INNER + 2*D_STATE + NHEADS
constexpr int kML = kB * kL; // 8192 rows
constexpr float kEPS = 1e-5f;

// ---- async copy of 32 contiguous bytes global -> LDS (2 x b128 per lane) ----
// INST_OFFSET is added to BOTH the LDS and the global address (ISA 08 §4.4),
// so a single (lds,vaddr) pair covers both halves.
__device__ __forceinline__ void async_cp32(const __bf16* g, __bf16* l) {
  unsigned lds = (unsigned)(uintptr_t)l;             // low 32 bits = LDS offset
  unsigned long long ga = (unsigned long long)(uintptr_t)g;
  asm volatile(
      "global_load_async_to_lds_b128 %0, %1, off\n\t"
      "global_load_async_to_lds_b128 %0, %1, off offset:16"
      :: "v"(lds), "v"(ga) : "memory");
}

// ---------------- fp32 -> bf16 convert ----------------
__global__ __launch_bounds__(256)
void k_f32_to_bf16(const float* __restrict__ s, __bf16* __restrict__ d, int n) {
  int i = blockIdx.x * 256 + threadIdx.x;
  if (i < n) d[i] = (__bf16)s[i];
}

// ---------------- LayerNorm -> bf16 ----------------
__global__ __launch_bounds__(256)
void k_layernorm_bf16(const float* __restrict__ x, const float* __restrict__ w,
                      const float* __restrict__ b, __bf16* __restrict__ h) {
  int row = blockIdx.x;          // 0..kML-1
  int tid = threadIdx.x;
  const float* xr = x + (size_t)row * kDM;
  float v[4];
  float s = 0.f, s2 = 0.f;
#pragma unroll
  for (int j = 0; j < 4; ++j) {
    v[j] = xr[tid + 256 * j];
    s  += v[j];
    s2 += v[j] * v[j];
  }
  __shared__ float r1[256], r2[256];
  r1[tid] = s; r2[tid] = s2;
  __syncthreads();
  for (int o = 128; o > 0; o >>= 1) {
    if (tid < o) { r1[tid] += r1[tid + o]; r2[tid] += r2[tid + o]; }
    __syncthreads();
  }
  float mu  = r1[0] * (1.f / kDM);
  float var = r2[0] * (1.f / kDM) - mu * mu;
  float inv = rsqrtf(var + kEPS);
#pragma unroll
  for (int j = 0; j < 4; ++j) {
    int c = tid + 256 * j;
    h[(size_t)row * kDM + c] = (__bf16)((v[j] - mu) * inv * w[c] + b[c]);
  }
}

// ---------------- bf16 WMMA GEMM:  C[M,N] = A[M,K] * W[N,K]^T ----------------
// Block tile 128x256, K-step 32. 8 waves: 2 along M x 4 along N, each wave
// owns a 64x64 output tile = 4x4 WMMA 16x16 tiles (16 WMMAs per K-step per
// wave, 8 LDS fragment loads -> high math-per-LDS-byte).
// Tiles are staged with GLOBAL_LOAD_ASYNC_TO_LDS_B128 into double-buffered LDS
// so the next K-tile's copies overlap the current tile's WMMAs (ASYNCcnt).
__global__ __launch_bounds__(256)
void k_gemm_bf16_wmma(const __bf16* __restrict__ A, const __bf16* __restrict__ W,
                      float* __restrict__ C, int M, int N, int K, int accum) {
  constexpr int BM = 128, BN = 256, BK = 32;
  __shared__ __bf16 la[2][BM * BK];   // 2 x 8 KB
  __shared__ __bf16 lb[2][BN * BK];   // 2 x 16 KB
  const int tid  = threadIdx.x;
  const int bm   = blockIdx.x * BM;
  const int bn   = blockIdx.y * BN;
  const int lane = tid & 31;
  const int wid  = tid >> 5;
  const int wm   = (wid & 1) * 64;   // wave M offset inside block tile
  const int wn   = (wid >> 1) * 64;  // wave N offset inside block tile
  const int lr   = tid >> 1;         // staging row (0..127)
  const int lc   = (tid & 1) * 16;   // 16-element K segment (0 or 16)

  // Clamp W rows at the N edge: rows >= N only feed output columns >= N,
  // which are masked at the store, so duplicated data is harmless.
  int wr0 = bn + lr;        if (wr0 >= N) wr0 = N - 1;
  int wr1 = bn + lr + 128;  if (wr1 >= N) wr1 = N - 1;
  const __bf16* gA  = A + (size_t)(bm + lr) * K + lc;
  const __bf16* gW0 = W + (size_t)wr0 * K + lc;
  const __bf16* gW1 = W + (size_t)wr1 * K + lc;

  v8f acc[4][4];
#pragma unroll
  for (int m = 0; m < 4; ++m)
#pragma unroll
    for (int n = 0; n < 4; ++n)
#pragma unroll
      for (int r = 0; r < 8; ++r) acc[m][n][r] = 0.f;

  const int arow = lane & 15;        // row/col within 16x16 fragment
  const int akc  = (lane >> 4) * 16; // K-half selected by lane hi bit

  // Prologue: stage K-tile 0 into buffer 0 (6 async b128 per thread).
  async_cp32(gA,  &la[0][lr * BK + lc]);
  async_cp32(gW0, &lb[0][lr * BK + lc]);
  async_cp32(gW1, &lb[0][(lr + 128) * BK + lc]);

  const int nIter = K / BK;
  for (int it = 0; it < nIter; ++it) {
    const int cur = it & 1;
    __syncthreads();                 // all waves done reading buffer cur^1
    if (it + 1 < nIter) {
      const size_t ko = (size_t)(it + 1) * BK;
      async_cp32(gA + ko,  &la[cur ^ 1][lr * BK + lc]);
      async_cp32(gW0 + ko, &lb[cur ^ 1][lr * BK + lc]);
      async_cp32(gW1 + ko, &lb[cur ^ 1][(lr + 128) * BK + lc]);
      if (it + 2 < nIter) {          // pull the tile after next into GL2
        __builtin_prefetch(gA + ko + BK, 0, 1);
        __builtin_prefetch(gW0 + ko + BK, 0, 1);
        __builtin_prefetch(gW1 + ko + BK, 0, 1);
      }
      asm volatile("s_wait_asynccnt 0x6" ::: "memory");  // tile `it` complete
    } else {
      asm volatile("s_wait_asynccnt 0x0" ::: "memory");
    }
    __syncthreads();                 // tile `it` visible to all waves

    v16bf bf[4];
#pragma unroll
    for (int n = 0; n < 4; ++n)
      bf[n] = *reinterpret_cast<const v16bf*>(&lb[cur][(wn + n * 16 + arow) * BK + akc]);
#pragma unroll
    for (int m = 0; m < 4; ++m) {
      v16bf af = *reinterpret_cast<const v16bf*>(&la[cur][(wm + m * 16 + arow) * BK + akc]);
#pragma unroll
      for (int n = 0; n < 4; ++n)
        acc[m][n] = __builtin_amdgcn_wmma_f32_16x16x32_bf16(
            false, af, false, bf[n], (short)0, acc[m][n], false, false);
    }
  }

  // Store: VGPR r holds row (r + 8*laneHi), col = lane&15 within 16x16 tile.
  const int chi = (lane >> 4) * 8;
  const int cn  = lane & 15;
#pragma unroll
  for (int m = 0; m < 4; ++m) {
    int rbase = bm + wm + m * 16 + chi;
#pragma unroll
    for (int n = 0; n < 4; ++n) {
      int cbase = bn + wn + n * 16 + cn;
      if (cbase < N) {
#pragma unroll
        for (int r = 0; r < 8; ++r) {
          size_t o = (size_t)(rbase + r) * N + cbase;
          float val = acc[m][n][r];
          if (accum) C[o] += val; else C[o] = val;
        }
      }
    }
  }
}

// ---------------- depthwise causal conv(4) + SiLU ----------------
// dir==0: y[l] = sum_k x[l-3+k]*w[k]   (forward, causal)
// dir==1: y[l] = sum_k x[l+3-k]*w[k]   (equivalent to flip->conv->flip)
__global__ __launch_bounds__(256)
void k_conv_silu(const float* __restrict__ zx, const float* __restrict__ cw,
                 const float* __restrict__ cb, float* __restrict__ out, int dir) {
  long long idx = (long long)blockIdx.x * 256 + threadIdx.x;
  if (idx >= (long long)kML * kCD) return;
  int c  = (int)(idx % kCD);
  int bl = (int)(idx / kCD);
  int l = bl % kL, b = bl / kL;
  float acc = cb[c];
#pragma unroll
  for (int k = 0; k < kDC; ++k) {
    int src = dir ? (l + (kDC - 1) - k) : (l - (kDC - 1) + k);
    if (src >= 0 && src < kL)
      acc += zx[((size_t)(b * kL + src)) * kDIP + kDI + c] * cw[c * kDC + k];
  }
  out[idx] = acc / (1.f + expf(-acc));  // SiLU
}

// ---------------- sequential SSM scan (one block per (b, head)) ----------------
// 256 threads: p = tid&63 (headdim), sg = tid>>6 selects 16 of 64 states.
// State (64x64) lives in registers: 16 floats per thread.
__global__ __launch_bounds__(256)
void k_scan(const float* __restrict__ zx, const float* __restrict__ xbc,
            const float* __restrict__ dt_bias, const float* __restrict__ A_log,
            const float* __restrict__ Dp, float* __restrict__ y, int dir) {
  int bh = blockIdx.x;
  int b = bh / kNH, hH = bh % kNH;
  int tid = threadIdx.x;
  int p = tid & 63, sg = tid >> 6;
  __shared__ float sB[kDS], sC[kDS], red[256];
  float st[16];
#pragma unroll
  for (int j = 0; j < 16; ++j) st[j] = 0.f;
  const float negA = -expf(A_log[hH]);
  const float dtb  = dt_bias[hH];
  const float Dh   = Dp[hH];

  for (int t = 0; t < kL; ++t) {
    int l = dir ? (kL - 1 - t) : t;
    size_t rowz = (size_t)(b * kL + l) * kDIP;
    size_t rowx = (size_t)(b * kL + l) * kCD;
    __syncthreads();  // protect sB/sC/red from previous step
    if (tid < kDS)            sB[tid]        = xbc[rowx + kDI + tid];
    else if (tid < 2 * kDS)   sC[tid - kDS]  = xbc[rowx + kDI + kDS + (tid - kDS)];
    __syncthreads();

    float dtr = zx[rowz + kDI + kCD + hH] + dtb;                 // dt tail of zxbcdt
    float dtv = (dtr > 20.f) ? dtr : log1pf(expf(dtr));          // softplus
    float dA  = expf(negA * dtv);
    float xv  = xbc[rowx + hH * kHD + p];
    float xdt = xv * dtv;
    float part = 0.f;
#pragma unroll
    for (int j = 0; j < 16; ++j) {
      int s = sg * 16 + j;
      st[j] = st[j] * dA + xdt * sB[s];
      part += st[j] * sC[s];
    }
    red[tid] = part;
    __syncthreads();
    if (sg == 0) {
      float yv = red[p] + red[64 + p] + red[128 + p] + red[192 + p] + Dh * xv;
      y[((size_t)(b * kL + l)) * kDI + hH * kHD + p] = yv;
    }
  }
}

// ---------------- gate (y * silu(z)) + RMSNorm -> bf16 ----------------
__global__ __launch_bounds__(256)
void k_gate_rms_bf16(const float* __restrict__ y, const float* __restrict__ zx,
                     const float* __restrict__ rw, __bf16* __restrict__ out) {
  int row = blockIdx.x;
  int tid = threadIdx.x;
  const float* yr = y + (size_t)row * kDI;
  const float* zr = zx + (size_t)row * kDIP;   // z = first kDI columns of zxbcdt
  float vals[8];
  float s2 = 0.f;
#pragma unroll
  for (int j = 0; j < 8; ++j) {
    int c = tid + 256 * j;
    float z = zr[c];
    float g = yr[c] * (z / (1.f + expf(-z)));
    vals[j] = g;
    s2 += g * g;
  }
  __shared__ float r1[256];
  r1[tid] = s2;
  __syncthreads();
  for (int o = 128; o > 0; o >>= 1) {
    if (tid < o) r1[tid] += r1[tid + o];
    __syncthreads();
  }
  float inv = rsqrtf(r1[0] * (1.f / kDI) + kEPS);
#pragma unroll
  for (int j = 0; j < 8; ++j) {
    int c = tid + 256 * j;
    out[(size_t)row * kDI + c] = (__bf16)(vals[j] * inv * rw[c]);
  }
}

// ---------------- final residual add ----------------
__global__ __launch_bounds__(256)
void k_final_add(const float* __restrict__ x, const float* __restrict__ o,
                 float* __restrict__ out, int n) {
  int i = blockIdx.x * 256 + threadIdx.x;
  if (i < n) out[i] = x[i] + o[i];
}

extern "C" void kernel_launch(void* const* d_in, const int* in_sizes, int n_in,
                              void* d_out, int out_size, void* d_ws, size_t ws_size,
                              hipStream_t stream) {
  (void)in_sizes; (void)n_in; (void)out_size; (void)ws_size;
  const float* x    = (const float*)d_in[0];
  const float* ln_w = (const float*)d_in[1];
  const float* ln_b = (const float*)d_in[2];
  const float* p_in_proj[2]  = {(const float*)d_in[3],  (const float*)d_in[11]};
  const float* p_conv_w[2]   = {(const float*)d_in[4],  (const float*)d_in[12]};
  const float* p_conv_b[2]   = {(const float*)d_in[5],  (const float*)d_in[13]};
  const float* p_dt_bias[2]  = {(const float*)d_in[6],  (const float*)d_in[14]};
  const float* p_A_log[2]    = {(const float*)d_in[7],  (const float*)d_in[15]};
  const float* p_D[2]        = {(const float*)d_in[8],  (const float*)d_in[16]};
  const float* p_rms_w[2]    = {(const float*)d_in[9],  (const float*)d_in[17]};
  const float* p_out_proj[2] = {(const float*)d_in[10], (const float*)d_in[18]};

  char* ws = (char*)d_ws;
  size_t off = 0;
  auto walloc = [&](size_t bytes) -> void* {
    void* p = ws + off;
    off += (bytes + 255) & ~(size_t)255;
    return p;
  };

  __bf16* h_bf      = (__bf16*)walloc((size_t)kML * kDM * 2);
  __bf16* wi_bf[2]  = {(__bf16*)walloc((size_t)kDIP * kDM * 2),
                       (__bf16*)walloc((size_t)kDIP * kDM * 2)};
  __bf16* wo_bf[2]  = {(__bf16*)walloc((size_t)kDM * kDI * 2),
                       (__bf16*)walloc((size_t)kDM * kDI * 2)};
  float*  zx        = (float*)walloc((size_t)kML * kDIP * 4);
  float*  xbc       = (float*)walloc((size_t)kML * kCD * 4);
  float*  ybuf      = (float*)walloc((size_t)kML * kDI * 4);
  __bf16* ybf       = (__bf16*)walloc((size_t)kML * kDI * 2);
  float*  oacc      = (float*)walloc((size_t)kML * kDM * 4);

  // Weight downconversion to bf16 (done every call; deterministic)
  const int n_wi = kDIP * kDM;
  const int n_wo = kDM * kDI;
  k_f32_to_bf16<<<(n_wi + 255) / 256, 256, 0, stream>>>(p_in_proj[0],  wi_bf[0], n_wi);
  k_f32_to_bf16<<<(n_wi + 255) / 256, 256, 0, stream>>>(p_in_proj[1],  wi_bf[1], n_wi);
  k_f32_to_bf16<<<(n_wo + 255) / 256, 256, 0, stream>>>(p_out_proj[0], wo_bf[0], n_wo);
  k_f32_to_bf16<<<(n_wo + 255) / 256, 256, 0, stream>>>(p_out_proj[1], wo_bf[1], n_wo);

  // LayerNorm -> bf16 activations
  k_layernorm_bf16<<<kML, 256, 0, stream>>>(x, ln_w, ln_b, h_bf);

  for (int dir = 0; dir < 2; ++dir) {
    // in_proj GEMM: (8192 x 4256) = h(8192x1024) * Wi(4256x1024)^T
    dim3 g1(kML / 128, (kDIP + 255) / 256);
    k_gemm_bf16_wmma<<<g1, 256, 0, stream>>>(h_bf, wi_bf[dir], zx, kML, kDIP, kDM, 0);

    // depthwise conv + SiLU
    long long ce = (long long)kML * kCD;
    k_conv_silu<<<(unsigned)((ce + 255) / 256), 256, 0, stream>>>(
        zx, p_conv_w[dir], p_conv_b[dir], xbc, dir);

    // sequential SSM scan
    k_scan<<<kB * kNH, 256, 0, stream>>>(zx, xbc, p_dt_bias[dir], p_A_log[dir],
                                         p_D[dir], ybuf, dir);

    // gate + RMSNorm -> bf16
    k_gate_rms_bf16<<<kML, 256, 0, stream>>>(ybuf, zx, p_rms_w[dir], ybf);

    // out_proj GEMM: (8192 x 1024) = y(8192x2048) * Wo(1024x2048)^T
    // dir 0 stores, dir 1 accumulates.
    dim3 g2(kML / 128, kDM / 256);
    k_gemm_bf16_wmma<<<g2, 256, 0, stream>>>(ybf, wo_bf[dir], oacc, kML, kDM, kDI, dir);
  }

  // out = x + fwd + rev
  k_final_add<<<(kML * kDM) / 256, 256, 0, stream>>>(x, oacc, (float*)d_out, kML * kDM);
}
